// StandardRouter_13761075216637
// MI455X (gfx1250) — compile-verified, matrix operand order checked by reading, not run on hardware
//
#include <hip/hip_runtime.h>
#include <hip/hip_bf16.h>
#include <stdint.h>

// ---------------------------------------------------------------------------
// MoE router for MI455X (gfx1250, wave32).
// - logits GEMM is HBM-bound (128MB fp32 activations @ 23.3TB/s ~ 5.5us);
//   matrix math uses v_wmma_f32_16x16x32_bf16 with a hi/lo bf16 split
//   (3 WMMAs per tile) for ~fp32 accuracy at bf16-WMMA throughput.
// - gate (B) operands are staged per-block into LDS with the CDNA5 async
//   LDS-DMA path (global_load_async_to_lds_b128 + s_wait_asynccnt), double
//   buffered, cutting B-side L2 traffic 8x (512MB -> 64MB).
// ---------------------------------------------------------------------------

typedef __attribute__((ext_vector_type(16))) __bf16        v16bf;
typedef __attribute__((ext_vector_type(8)))  float         v8f;
typedef __attribute__((ext_vector_type(8)))  unsigned int  v8u;

#define NUM_TOKENS   8192
#define DMODEL       4096
#define NEXPERTS     64
#define TOPK         8
#define TOK_PER_BLK  128
#define NBLOCKS      (NUM_TOKENS / TOK_PER_BLK)   // 64
#define NCHUNK       (DMODEL / 32)                // 128 K-chunks
#define LDS_STRIDE   65                           // 64 + 1 pad vs bank conflicts
#define BBUF_BYTES   8192                         // per buffer: hi 4KB + lo 4KB

// Split two fp32 into packed bf16-hi pair and bf16-lo (residual) pair.
// uint layout matches WMMA 16-bit operand: first element in [15:0].
__device__ __forceinline__ uint2 split2(float f0, float f1) {
    unsigned int b0 = __float_as_uint(f0), b1 = __float_as_uint(f1);
    unsigned int h0 = b0 & 0xFFFF0000u,    h1 = b1 & 0xFFFF0000u;
    float l0 = f0 - __uint_as_float(h0);
    float l1 = f1 - __uint_as_float(h1);
    unsigned int lb0 = __float_as_uint(l0) & 0xFFFF0000u;
    unsigned int lb1 = __float_as_uint(l1) & 0xFFFF0000u;
    uint2 r;
    r.x = (h0 >> 16) | h1;      // hi pair
    r.y = (lb0 >> 16) | lb1;    // lo pair
    return r;
}

// CDNA5 async LDS DMA: move 16B from global memory to LDS, tracked by ASYNCcnt.
__device__ __forceinline__ void async_ld16(unsigned lds_off, const void* gaddr) {
    asm volatile("global_load_async_to_lds_b128 %0, %1, off"
                 :: "v"(lds_off), "v"((unsigned long long)(size_t)gaddr)
                 : "memory");
}

// ---------------------------------------------------------------------------
// Kernel 0: split gate_w [64,4096] fp32 -> hi/lo bf16 arrays (same layout).
// ---------------------------------------------------------------------------
__global__ void prep_gate(const float* __restrict__ gw,
                          unsigned short* __restrict__ ghi,
                          unsigned short* __restrict__ glo) {
    int i = blockIdx.x * 256 + threadIdx.x;            // exact grid, no bounds
    float f = gw[i];
    unsigned int b = __float_as_uint(f);
    unsigned int h = b & 0xFFFF0000u;
    ghi[i] = (unsigned short)(h >> 16);
    float lo = f - __uint_as_float(h);
    glo[i] = (unsigned short)(__float_as_uint(lo) >> 16);
}

// ---------------------------------------------------------------------------
// Kernel 1: GEMM (split-bf16 WMMA, async-LDS staged B) + softmax + top-8 +
// per-block prob sums. Block: 256 threads = 8 waves; each wave: 16 tokens.
// ---------------------------------------------------------------------------
__global__ __launch_bounds__(256) void router_main(
    const float* __restrict__ hidden,
    const unsigned short* __restrict__ ghi,
    const unsigned short* __restrict__ glo,
    float* __restrict__ partials,      // [NBLOCKS][NEXPERTS]
    float* __restrict__ out_idx,       // [NUM_TOKENS][TOPK] (indices as float)
    float* __restrict__ out_w)         // [NUM_TOKENS][TOPK]
{
    // Double-buffered B staging: [buf][hi|lo][expert 0..63][64B = 32 bf16]
    __shared__ __attribute__((aligned(16))) unsigned char smB[2 * BBUF_BYTES];
    __shared__ float sm[TOK_PER_BLK * LDS_STRIDE];

    const int tid  = threadIdx.x;
    const int wave = tid >> 5;
    const int lane = tid & 31;
    const int g    = lane >> 4;        // half-wave group (K interleave)
    const int li   = lane & 15;
    const int tokBase = blockIdx.x * TOK_PER_BLK;

    // ---- staging assignment: thread tid moves 16B of hi and 16B of lo per
    // chunk; row = tid>>2 (expert), seg = tid&3 (16B segment of 64B row).
    const int srow = tid >> 2, sseg = tid & 3;
    const unsigned short* gsrc_hi = ghi + (size_t)srow * DMODEL + sseg * 8;
    const unsigned short* gsrc_lo = glo + (size_t)srow * DMODEL + sseg * 8;
    const unsigned lds_hi0 = (unsigned)(size_t)&smB[0] + (unsigned)tid * 16u;
    const unsigned lds_lo0 = lds_hi0 + 4096u;

    // WMMA 16-bit operand layout: lanes 0-15 hold K = k0+[0..7] and k0+[16..23],
    // lanes 16-31 hold K = k0+[8..15] and k0+[24..31].
    const float* arow = hidden + (size_t)(tokBase + wave * 16 + li) * DMODEL;
    const int kg  = 8 * g;             // K sub-offset for this half-wave
    const unsigned char* smB_rd = smB + (size_t)li * 64 + (size_t)kg * 2;

    v8f acc[4];
    #pragma unroll
    for (int t = 0; t < 4; ++t) acc[t] = (v8f)0.0f;

    // prologue: stage chunk 0 into buffer 0
    async_ld16(lds_hi0, gsrc_hi);
    async_ld16(lds_lo0, gsrc_lo);

    for (int i = 0; i < NCHUNK; ++i) {
        const int k0  = i * 32;
        const unsigned boff = (i & 1) ? (unsigned)BBUF_BYTES : 0u;

        __syncthreads();  // all readers of the other buffer are done
        if (i + 1 < NCHUNK) {
            const unsigned nboff = boff ^ (unsigned)BBUF_BYTES;
            async_ld16(lds_hi0 + nboff, gsrc_hi + (size_t)(i + 1) * 32);
            async_ld16(lds_lo0 + nboff, gsrc_lo + (size_t)(i + 1) * 32);
            asm volatile("s_wait_asynccnt 0x2" ::: "memory");  // chunk i landed
        } else {
            asm volatile("s_wait_asynccnt 0x0" ::: "memory");
        }
        __syncthreads();  // chunk i visible from every wave's DMA

        // ---- A: stream fp32 activations from HBM, split to hi/lo bf16
        const int ka = k0 + kg;
        const int kb = ka + 16;
        __builtin_prefetch(arow + k0 + 128, 0, 3);   // global_prefetch_b8

        float4 a0 = *(const float4*)(arow + ka);
        float4 a1 = *(const float4*)(arow + ka + 4);
        float4 a2 = *(const float4*)(arow + kb);
        float4 a3 = *(const float4*)(arow + kb + 4);

        v8u hu, lu;
        { uint2 r = split2(a0.x, a0.y); hu[0] = r.x; lu[0] = r.y; }
        { uint2 r = split2(a0.z, a0.w); hu[1] = r.x; lu[1] = r.y; }
        { uint2 r = split2(a1.x, a1.y); hu[2] = r.x; lu[2] = r.y; }
        { uint2 r = split2(a1.z, a1.w); hu[3] = r.x; lu[3] = r.y; }
        { uint2 r = split2(a2.x, a2.y); hu[4] = r.x; lu[4] = r.y; }
        { uint2 r = split2(a2.z, a2.w); hu[5] = r.x; lu[5] = r.y; }
        { uint2 r = split2(a3.x, a3.y); hu[6] = r.x; lu[6] = r.y; }
        { uint2 r = split2(a3.z, a3.w); hu[7] = r.x; lu[7] = r.y; }
        v16bf Ahi = __builtin_bit_cast(v16bf, hu);
        v16bf Alo = __builtin_bit_cast(v16bf, lu);

        // ---- B: read staged hi/lo operands from LDS (ds_load_b128)
        #pragma unroll
        for (int t = 0; t < 4; ++t) {
            const unsigned char* bh = smB_rd + boff + (size_t)(t * 16) * 64;
            const unsigned char* bl = bh + 4096;
            uint4 h0 = *(const uint4*)(bh);
            uint4 h1 = *(const uint4*)(bh + 32);
            uint4 l0 = *(const uint4*)(bl);
            uint4 l1 = *(const uint4*)(bl + 32);
            v8u bhv, blv;
            bhv[0]=h0.x; bhv[1]=h0.y; bhv[2]=h0.z; bhv[3]=h0.w;
            bhv[4]=h1.x; bhv[5]=h1.y; bhv[6]=h1.z; bhv[7]=h1.w;
            blv[0]=l0.x; blv[1]=l0.y; blv[2]=l0.z; blv[3]=l0.w;
            blv[4]=l1.x; blv[5]=l1.y; blv[6]=l1.z; blv[7]=l1.w;
            v16bf Bhi = __builtin_bit_cast(v16bf, bhv);
            v16bf Blo = __builtin_bit_cast(v16bf, blv);
            // (Ahi+Alo)(Bhi+Blo) ~ Ahi*Bhi + Ahi*Blo + Alo*Bhi  (drop lo*lo)
            acc[t] = __builtin_amdgcn_wmma_f32_16x16x32_bf16(
                         false, Ahi, false, Bhi, (short)0, acc[t], false, false);
            acc[t] = __builtin_amdgcn_wmma_f32_16x16x32_bf16(
                         false, Ahi, false, Blo, (short)0, acc[t], false, false);
            acc[t] = __builtin_amdgcn_wmma_f32_16x16x32_bf16(
                         false, Alo, false, Bhi, (short)0, acc[t], false, false);
        }
    }

    // Spill logits to LDS. C/D layout: VGPR r <-> row (r + 8*g), col = lane%16.
    #pragma unroll
    for (int t = 0; t < 4; ++t) {
        const int trow = wave * 16 + 8 * g;
        const int col  = t * 16 + li;
        #pragma unroll
        for (int r = 0; r < 8; ++r)
            sm[(trow + r) * LDS_STRIDE + col] = acc[t][r];
    }
    __syncthreads();

    // One thread per token: softmax + stable top-8 (value desc, index asc).
    if (tid < TOK_PER_BLK) {
        float v[NEXPERTS];
        #pragma unroll
        for (int e = 0; e < NEXPERTS; ++e) v[e] = sm[tid * LDS_STRIDE + e];

        float mx = v[0];
        #pragma unroll
        for (int e = 1; e < NEXPERTS; ++e) mx = fmaxf(mx, v[e]);
        float S = 0.0f;
        #pragma unroll
        for (int e = 0; e < NEXPERTS; ++e) { float ex = __expf(v[e] - mx); v[e] = ex; S += ex; }
        const float invS = 1.0f / S;

        // full softmax probs back to LDS for the aux-loss partial sums
        #pragma unroll
        for (int e = 0; e < NEXPERTS; ++e) sm[tid * LDS_STRIDE + e] = v[e] * invS;

        // top-8 without dynamic register indexing: eligibility scan
        float prevB = 3.0e38f; int prevI = -1;
        float wsel[TOPK]; int isel[TOPK];
        float wsum = 0.0f;
        #pragma unroll
        for (int j = 0; j < TOPK; ++j) {
            float best = -1.0f; int bi = 0;
            #pragma unroll
            for (int e = 0; e < NEXPERTS; ++e) {
                bool elig = (v[e] < prevB) || (v[e] == prevB && e > prevI);
                bool take = elig && (v[e] > best);
                best = take ? v[e] : best;
                bi   = take ? e    : bi;
            }
            wsel[j] = best; isel[j] = bi;
            prevB = best; prevI = bi;
            wsum += best;
        }
        const float invW = 1.0f / wsum;   // exp-ratio == prob-ratio
        const int tok = tokBase + tid;
        #pragma unroll
        for (int j = 0; j < TOPK; ++j) {
            out_idx[(size_t)tok * TOPK + j] = (float)isel[j];
            out_w  [(size_t)tok * TOPK + j] = wsel[j] * invW;
        }
    }
    __syncthreads();

    // Deterministic per-block expert prob sums (fixed accumulation order).
    if (tid < NEXPERTS) {
        float s = 0.0f;
        #pragma unroll 4
        for (int r = 0; r < TOK_PER_BLK; ++r) s += sm[r * LDS_STRIDE + tid];
        partials[blockIdx.x * NEXPERTS + tid] = s;
    }
}

// ---------------------------------------------------------------------------
// Kernel 2: aux loss = E * sum(mean_prob^2) * 0.001, fixed-order reduction.
// ---------------------------------------------------------------------------
__global__ void router_aux(const float* __restrict__ partials,
                           float* __restrict__ aux_out) {
    __shared__ float sq[NEXPERTS];
    const int e = threadIdx.x;
    float s = 0.0f;
    #pragma unroll
    for (int b = 0; b < NBLOCKS; ++b) s += partials[b * NEXPERTS + e];
    float avg = s * (1.0f / (float)NUM_TOKENS);
    sq[e] = avg * avg;
    __syncthreads();
    if (e == 0) {
        float t = 0.0f;
        #pragma unroll
        for (int i = 0; i < NEXPERTS; ++i) t += sq[i];
        aux_out[0] = (float)NEXPERTS * t * 0.001f;
    }
}

// ---------------------------------------------------------------------------
extern "C" void kernel_launch(void* const* d_in, const int* in_sizes, int n_in,
                              void* d_out, int out_size, void* d_ws, size_t ws_size,
                              hipStream_t stream) {
    (void)in_sizes; (void)n_in; (void)out_size; (void)ws_size;

    const float* hidden = (const float*)d_in[0];   // [8192, 4096] fp32
    const float* gate_w = (const float*)d_in[1];   // [64, 4096]   fp32

    char* ws = (char*)d_ws;
    unsigned short* ghi = (unsigned short*)ws;                                   // 512 KB
    unsigned short* glo = (unsigned short*)(ws + (size_t)NEXPERTS * DMODEL * 2); // 512 KB
    float* partials = (float*)(ws + (size_t)NEXPERTS * DMODEL * 4);              // 16 KB

    float* out      = (float*)d_out;
    float* out_idx  = out;                                     // [8192,8]
    float* out_w    = out + (size_t)NUM_TOKENS * TOPK;         // [8192,8]
    float* aux_out  = out + (size_t)2 * NUM_TOKENS * TOPK;     // [1]

    prep_gate  <<<(NEXPERTS * DMODEL) / 256, 256, 0, stream>>>(gate_w, ghi, glo);
    router_main<<<NBLOCKS, 256, 0, stream>>>(hidden, ghi, glo, partials, out_idx, out_w);
    router_aux <<<1, NEXPERTS, 0, stream>>>(partials, aux_out);
}
